// PlanningNetwork_73194832659235
// MI455X (gfx1250) — matrix operand order repaired and checked by reading, not compiled, not run on hardware
//
#include <hip/hip_runtime.h>
#include <hip/hip_bf16.h>

// ---------------------------------------------------------------------------
// PlanningNetwork for MI455X (gfx1250), wave32 + WMMA bf16.
//
// Roofline: LSTM GEMMs = ~275 GFLOP (dominant), output traffic 273 MB
// (~12 us @ 23.3 TB/s). Compute-bound -> bf16 WMMA (v_wmma_f32_16x16x32_bf16,
// f32 accumulate). Weights (~1.3 MB) stay L2/WGP$-resident. LSTM:
//  - 64-row batch tile per block, 16 waves = 4 row-slabs x 4 column-quarters
//    (2 output tiles per wave -> small unrolled body, low VGPR pressure)
//  - h1/h2 double-buffered bf16 in LDS (80 KB/block) for the lane transpose
//  - c1/c2 live entirely in VGPRs (fragment layout == gate accumulator)
//  - bias folded into WMMA accumulator init; native CDNA5 TANH trans op
// ---------------------------------------------------------------------------

#define BB 16384
#define HH 128
#define TT 32

typedef __attribute__((ext_vector_type(16))) __bf16 v16bf;
typedef __attribute__((ext_vector_type(8)))  float  v8f;
typedef unsigned short u16;
typedef unsigned int   u32;

__device__ __forceinline__ u16 f2bf(float f) {               // RNE f32 -> bf16
  u32 x = __float_as_uint(f);
  return (u16)((x + 0x7FFFu + ((x >> 16) & 1u)) >> 16);
}
__device__ __forceinline__ float bf2f(u16 h) {
  return __uint_as_float(((u32)h) << 16);
}

__device__ __forceinline__ float fast_tanh(float x) {
#if __has_builtin(__builtin_amdgcn_tanhf)
  return __builtin_amdgcn_tanhf(x);          // native v_tanh_f32 (CDNA5 TRANS)
#else
  return tanhf(x);
#endif
}
// exact identity: sigmoid(x) = 0.5*tanh(x/2) + 0.5  -> 1 TRANS + 2 VALU
__device__ __forceinline__ float sigm(float x) {
  return fmaf(0.5f, fast_tanh(0.5f * x), 0.5f);
}
__device__ __forceinline__ v8f splat8(float x) {
  v8f v = {x, x, x, x, x, x, x, x};
  return v;
}

union Frag { v16bf v; uint4 q[2]; };

// A fragment (16x32 bf16 tile, rows = wave's 16 batch rows) from a row-major
// bf16 buffer. ISA 16-bit A layout: lanes 0-15: M=lane, K={0..7,16..23};
// lanes 16-31: M=lane-16, K={8..15,24..31}. All chunks 16B-aligned.
__device__ __forceinline__ v16bf load_a(const u16* base, int row0, int ld, int kc, int lane) {
  const int m = lane & 15, hi = lane >> 4;
  const u16* p = base + (size_t)(row0 + m) * ld + kc * 32 + hi * 8;
  Frag f;
  f.q[0] = *(const uint4*)(p);
  f.q[1] = *(const uint4*)(p + 16);
  return f.v;
}

// B fragment (32x16): B[k][n] = W[n][k] with W row-major (out x in) bf16.
// Lanes 0-15: N=lane, K={0..7,16..23}; lanes 16-31: N=lane-16, K={8..15,24..31}.
__device__ __forceinline__ v16bf load_b(const u16* W, int ld, int n0, int kc, int lane) {
  const int n = lane & 15, hi = lane >> 4;
  const u16* p = W + (size_t)(n0 + n) * ld + kc * 32 + hi * 8;
  Frag f;
  f.q[0] = *(const uint4*)(p);
  f.q[1] = *(const uint4*)(p + 16);
  return f.v;
}

__device__ __forceinline__ v8f wmma_bf(v16bf a, v16bf b, v8f c) {
  return __builtin_amdgcn_wmma_f32_16x16x32_bf16(false, a, false, b, (short)0, c,
                                                 false, false);
}

// ---------------------------------------------------------------------------
// Weight prep
// ---------------------------------------------------------------------------
__global__ void convert_bf16_kernel(const float* __restrict__ src,
                                    u16* __restrict__ dst, int n) {
  int i = blockIdx.x * 256 + threadIdx.x;
  if (i < n) dst[i] = f2bf(src[i]);
}

__global__ void bias_sum_kernel(const float* __restrict__ a,
                                const float* __restrict__ b,
                                float* __restrict__ o, int n) {
  int i = blockIdx.x * 256 + threadIdx.x;
  if (i < n) o[i] = a[i] + b[i];
}

// ---------------------------------------------------------------------------
// Encoder: relu(X@W1^T+b1) -> relu(@W2^T+b2) -> ctx = @W3^T+b3
// 32 batch rows / block, 2 waves, each wave owns 16 rows.
// ---------------------------------------------------------------------------
__global__ __launch_bounds__(64) void encoder_kernel(
    const float* __restrict__ gc,
    const u16* __restrict__ wW1, const float* __restrict__ b1,
    const u16* __restrict__ wW2, const float* __restrict__ b2,
    const u16* __restrict__ wW3, const float* __restrict__ b3,
    float* __restrict__ out_ctx, u16* __restrict__ x0) {
  __shared__ __align__(16) u16 Xb[32 * 128];
  __shared__ __align__(16) u16 Y1b[32 * 256];
  __shared__ __align__(16) u16 Y2b[32 * 128];
  const int tid = threadIdx.x;
  const int lane = tid & 31, wid = tid >> 5;
  const int hi = lane >> 4, l15 = lane & 15;
  const int row0 = blockIdx.x * 32;

  for (int i = tid; i < 32 * 128; i += 64)
    Xb[i] = f2bf(gc[(size_t)(row0 + (i >> 7)) * 128 + (i & 127)]);
  __syncthreads();

  const int wr = wid * 16;
  // L1: 128 -> 256, relu
  {
    v16bf a[4];
#pragma unroll
    for (int kc = 0; kc < 4; ++kc) a[kc] = load_a(Xb, wr, 128, kc, lane);
    for (int ti = 0; ti < 16; ++ti) {
      const int n = ti * 16 + l15;
      v8f acc = splat8(b1[n]);                    // bias folded into C init
#pragma unroll
      for (int kc = 0; kc < 4; ++kc)
        acc = wmma_bf(a[kc], load_b(wW1, 128, ti * 16, kc, lane), acc);
#pragma unroll
      for (int r = 0; r < 8; ++r) {
        const int m = wr + r + hi * 8;
        Y1b[m * 256 + n] = f2bf(fmaxf(acc[r], 0.f));
      }
    }
  }
  // L2: 256 -> 128, relu  (own rows only -> no barrier needed)
  {
    v16bf a[8];
#pragma unroll
    for (int kc = 0; kc < 8; ++kc) a[kc] = load_a(Y1b, wr, 256, kc, lane);
    for (int ti = 0; ti < 8; ++ti) {
      const int n = ti * 16 + l15;
      v8f acc = splat8(b2[n]);
#pragma unroll
      for (int kc = 0; kc < 8; ++kc)
        acc = wmma_bf(a[kc], load_b(wW2, 256, ti * 16, kc, lane), acc);
#pragma unroll
      for (int r = 0; r < 8; ++r) {
        const int m = wr + r + hi * 8;
        Y2b[m * 128 + n] = f2bf(fmaxf(acc[r], 0.f));
      }
    }
  }
  // L3: 128 -> 64 (linear). Write ctx (f32 output) and zero-padded bf16 x0.
  {
    v16bf a[4];
#pragma unroll
    for (int kc = 0; kc < 4; ++kc) a[kc] = load_a(Y2b, wr, 128, kc, lane);
    for (int ti = 0; ti < 4; ++ti) {
      const int n = ti * 16 + l15;
      v8f acc = splat8(b3[n]);
#pragma unroll
      for (int kc = 0; kc < 4; ++kc)
        acc = wmma_bf(a[kc], load_b(wW3, 128, ti * 16, kc, lane), acc);
#pragma unroll
      for (int r = 0; r < 8; ++r) {
        const int m = wr + r + hi * 8;
        const float v = acc[r];
        out_ctx[(size_t)(row0 + m) * 64 + n] = v;
        x0[(size_t)(row0 + m) * 128 + n] = f2bf(v);
        x0[(size_t)(row0 + m) * 128 + 64 + n] = 0;  // zero pad to H=128
      }
    }
  }
}

// ---------------------------------------------------------------------------
// LSTM cell, one wave: 16 rows x NT output tiles (cols t0*16 ..).
// A-fragments preloaded by caller; c-state lives in registers (cst[NT], same
// fragment layout as the gate accumulators -> zero index math).
// ---------------------------------------------------------------------------
template <bool STORE_OUT, int NT>
__device__ __forceinline__ void lstm_cell(
    const v16bf* fx, const v16bf* fh,           // preloaded A frags [4]
    const u16* __restrict__ Wih, const u16* __restrict__ Whh,
    const float* __restrict__ bsum,
    v8f* cst,                                   // [NT] register c-state
    u16* Hout, float* gout,                     // gout: &seq[row0g][t][0]
    int wr, int t0, int lane) {
  const int hi = lane >> 4, l15 = lane & 15;
#pragma unroll
  for (int ti = 0; ti < NT; ++ti) {
    const int n0 = (t0 + ti) * 16;
    const int n = n0 + l15;
    v8f Ai = splat8(bsum[n]);
    v8f Af = splat8(bsum[128 + n]);
    v8f Ag = splat8(bsum[256 + n]);
    v8f Ao = splat8(bsum[384 + n]);
#pragma unroll
    for (int kc = 0; kc < 4; ++kc) {
      Ai = wmma_bf(fx[kc], load_b(Wih, 128, n0, kc, lane), Ai);
      Af = wmma_bf(fx[kc], load_b(Wih, 128, 128 + n0, kc, lane), Af);
      Ag = wmma_bf(fx[kc], load_b(Wih, 128, 256 + n0, kc, lane), Ag);
      Ao = wmma_bf(fx[kc], load_b(Wih, 128, 384 + n0, kc, lane), Ao);
      Ai = wmma_bf(fh[kc], load_b(Whh, 128, n0, kc, lane), Ai);
      Af = wmma_bf(fh[kc], load_b(Whh, 128, 128 + n0, kc, lane), Af);
      Ag = wmma_bf(fh[kc], load_b(Whh, 128, 256 + n0, kc, lane), Ag);
      Ao = wmma_bf(fh[kc], load_b(Whh, 128, 384 + n0, kc, lane), Ao);
    }
    v8f c = cst[ti];
#pragma unroll
    for (int r = 0; r < 8; ++r) {
      const int m = wr + r + hi * 8;
      const float iv = sigm(Ai[r]);
      const float fv = sigm(Af[r]);
      const float gv = fast_tanh(Ag[r]);
      const float ov = sigm(Ao[r]);
      const float cc = fv * c[r] + iv * gv;
      c[r] = cc;
      const float h = ov * fast_tanh(cc);
      Hout[m * 128 + n] = f2bf(h);
      if (STORE_OUT) gout[(size_t)m * (TT * 128) + n] = h;
    }
    cst[ti] = c;
  }
}

// 64 batch rows / block, 16 waves = 4 row-slabs x 4 column-quarters.
// 80 KB dynamic LDS; one barrier per cell (column-quarters exchange h).
__global__ __launch_bounds__(512) void lstm_kernel(
    const u16* __restrict__ x0,
    const u16* __restrict__ Wih1, const u16* __restrict__ Whh1,
    const u16* __restrict__ Wih2, const u16* __restrict__ Whh2,
    const float* __restrict__ bsum1, const float* __restrict__ bsum2,
    const int* __restrict__ seq_len,
    float* __restrict__ out_seq) {
  extern __shared__ __align__(16) char smem[];
  const int NB = 64 * 128;             // elements per h buffer
  u16* X0b = (u16*)smem;               // 64x128 bf16
  u16* H1  = X0b + NB;                 // 2 x 64x128 bf16 (double buffer)
  u16* H2  = H1 + 2 * NB;              // 2 x 64x128 bf16

  const int tid = threadIdx.x, lane = tid & 31, wid = tid >> 5;
  const int wr = (wid & 3) * 16;       // row slab
  const int t0 = (wid >> 2) * 2;       // first of this wave's 2 column tiles
  const int row0 = blockIdx.x * 64;

  for (int i = tid; i < NB; i += 512) {
    X0b[i] = x0[(size_t)(row0 + (i >> 7)) * 128 + (i & 127)];
    H1[i] = 0;                         // only cur=0 halves need zeroing
    H2[i] = 0;
  }
  __syncthreads();

  int T = *seq_len;
  if (T > TT) T = TT;
  if (T < 0) T = 0;

  v8f c1[2], c2[2];                    // register-resident cell state
#pragma unroll
  for (int ti = 0; ti < 2; ++ti) {
    c1[ti] = splat8(0.f);
    c2[ti] = splat8(0.f);
  }

  float* gbase = out_seq + (size_t)row0 * (TT * 128);
  int cur = 0;
  for (int t = 0; t < T; ++t) {
    v16bf fx[4], fh1[4], fh2[4];
    const u16* Hx = (t == 0) ? X0b : (H2 + cur * NB);   // x_t = h2_{t-1}
#pragma unroll
    for (int kc = 0; kc < 4; ++kc) {
      fx[kc]  = load_a(Hx, wr, 128, kc, lane);
      fh1[kc] = load_a(H1 + cur * NB, wr, 128, kc, lane);
    }
    if (t == 0) {                      // h2_0 = 0 (zeroed LDS), x0 differs
#pragma unroll
      for (int kc = 0; kc < 4; ++kc) fh2[kc] = load_a(H2, wr, 128, kc, lane);
    } else {                           // cell2 hidden == cell1 input buffer
#pragma unroll
      for (int kc = 0; kc < 4; ++kc) fh2[kc] = fx[kc];
    }
    lstm_cell<false, 2>(fx, fh1, Wih1, Whh1, bsum1, c1,
                        H1 + (cur ^ 1) * NB, nullptr, wr, t0, lane);
    __syncthreads();                   // h1 quarters exchanged between waves
    v16bf f1n[4];
#pragma unroll
    for (int kc = 0; kc < 4; ++kc)
      f1n[kc] = load_a(H1 + (cur ^ 1) * NB, wr, 128, kc, lane);
    lstm_cell<true, 2>(f1n, fh2, Wih2, Whh2, bsum2, c2,
                       H2 + (cur ^ 1) * NB, gbase + t * 128, wr, t0, lane);
    __syncthreads();                   // h2 quarters exchanged
    cur ^= 1;
  }
}

// ---------------------------------------------------------------------------
// Quality head: [h2_final, ctx] (192) -> 128 relu -> 64 relu -> 1 sigmoid
// ---------------------------------------------------------------------------
__global__ __launch_bounds__(64) void quality_kernel(
    const float* __restrict__ out_seq, const float* __restrict__ out_ctx,
    const u16* __restrict__ wWq1, const float* __restrict__ bq1,
    const u16* __restrict__ wWq2, const float* __restrict__ bq2,
    const float* __restrict__ Wq3, const float* __restrict__ bq3,
    const int* __restrict__ seq_len,
    float* __restrict__ out_q) {
  __shared__ __align__(16) u16 Qin[32 * 192];
  __shared__ __align__(16) u16 H1q[32 * 128];
  __shared__ __align__(16) u16 H2q[32 * 64];
  const int tid = threadIdx.x, lane = tid & 31, wid = tid >> 5;
  const int hi = lane >> 4, l15 = lane & 15;
  const int row0 = blockIdx.x * 32;

  int T = *seq_len;
  if (T > TT) T = TT;
  if (T < 1) T = 1;
  const int tlast = T - 1;

  for (int i = tid; i < 32 * 192; i += 64) {
    const int r = i / 192, c = i % 192;
    float v = (c < 128)
                  ? out_seq[(size_t)(row0 + r) * (TT * 128) + tlast * 128 + c]
                  : out_ctx[(size_t)(row0 + r) * 64 + (c - 128)];
    Qin[i] = f2bf(v);
  }
  __syncthreads();

  const int wr = wid * 16;
  // L1: 192 -> 128 relu
  {
    v16bf a[6];
#pragma unroll
    for (int kc = 0; kc < 6; ++kc) a[kc] = load_a(Qin, wr, 192, kc, lane);
    for (int ti = 0; ti < 8; ++ti) {
      const int n = ti * 16 + l15;
      v8f acc = splat8(bq1[n]);
#pragma unroll
      for (int kc = 0; kc < 6; ++kc)
        acc = wmma_bf(a[kc], load_b(wWq1, 192, ti * 16, kc, lane), acc);
#pragma unroll
      for (int r = 0; r < 8; ++r) {
        const int m = wr + r + hi * 8;
        H1q[m * 128 + n] = f2bf(fmaxf(acc[r], 0.f));
      }
    }
  }
  // L2: 128 -> 64 relu
  {
    v16bf a[4];
#pragma unroll
    for (int kc = 0; kc < 4; ++kc) a[kc] = load_a(H1q, wr, 128, kc, lane);
    for (int ti = 0; ti < 4; ++ti) {
      const int n = ti * 16 + l15;
      v8f acc = splat8(bq2[n]);
#pragma unroll
      for (int kc = 0; kc < 4; ++kc)
        acc = wmma_bf(a[kc], load_b(wWq2, 128, ti * 16, kc, lane), acc);
#pragma unroll
      for (int r = 0; r < 8; ++r) {
        const int m = wr + r + hi * 8;
        H2q[m * 64 + n] = f2bf(fmaxf(acc[r], 0.f));
      }
    }
  }
  __syncthreads();
  // L3: 64 -> 1 + sigmoid (one thread per row)
  if (tid < 32) {
    float s = bq3[0];
#pragma unroll
    for (int k = 0; k < 64; ++k) s += bf2f(H2q[tid * 64 + k]) * Wq3[k];
    out_q[row0 + tid] = sigm(s);
  }
}

// ---------------------------------------------------------------------------
// Small heads from ctx: plan-type (softmax), duration, cost. ~0.2 GFLOP total.
// ---------------------------------------------------------------------------
__global__ __launch_bounds__(256) void heads_kernel(
    const float* __restrict__ ctx,
    const float* __restrict__ Wp1, const float* __restrict__ bp1,
    const float* __restrict__ Wp2, const float* __restrict__ bp2,
    const float* __restrict__ Wd1, const float* __restrict__ bd1,
    const float* __restrict__ Wd2, const float* __restrict__ bd2,
    const float* __restrict__ Wc1, const float* __restrict__ bc1,
    const float* __restrict__ Wc2, const float* __restrict__ bc2,
    float* __restrict__ out_p, float* __restrict__ out_d,
    float* __restrict__ out_c) {
  const int b = blockIdx.x * 256 + threadIdx.x;
  if (b >= BB) return;
  float cx[64];
#pragma unroll
  for (int k = 0; k < 64; ++k) cx[k] = ctx[(size_t)b * 64 + k];

  // plan type: 64 -> 32 relu -> 8 softmax
  float hp[32];
#pragma unroll 4
  for (int j = 0; j < 32; ++j) {
    float s = bp1[j];
    for (int k = 0; k < 64; ++k) s += Wp1[j * 64 + k] * cx[k];
    hp[j] = fmaxf(s, 0.f);
  }
  float lg[8], mx = -1e30f;
#pragma unroll
  for (int j = 0; j < 8; ++j) {
    float s = bp2[j];
    for (int k = 0; k < 32; ++k) s += Wp2[j * 32 + k] * hp[k];
    lg[j] = s;
    mx = fmaxf(mx, s);
  }
  float se = 0.f;
#pragma unroll
  for (int j = 0; j < 8; ++j) { lg[j] = __expf(lg[j] - mx); se += lg[j]; }
  const float inv = 1.f / se;
#pragma unroll
  for (int j = 0; j < 8; ++j) out_p[(size_t)b * 8 + j] = lg[j] * inv;

  // duration: relu(relu(ctx@Wd1^T+bd1)@Wd2^T+bd2)
  float s2 = bd2[0];
#pragma unroll 4
  for (int j = 0; j < 32; ++j) {
    float s = bd1[j];
    for (int k = 0; k < 64; ++k) s += Wd1[j * 64 + k] * cx[k];
    s2 += Wd2[j] * fmaxf(s, 0.f);
  }
  out_d[b] = fmaxf(s2, 0.f);

  // cost
  float s3 = bc2[0];
#pragma unroll 4
  for (int j = 0; j < 32; ++j) {
    float s = bc1[j];
    for (int k = 0; k < 64; ++k) s += Wc1[j * 64 + k] * cx[k];
    s3 += Wc2[j] * fmaxf(s, 0.f);
  }
  out_c[b] = fmaxf(s3, 0.f);
}

// ---------------------------------------------------------------------------
extern "C" void kernel_launch(void* const* d_in, const int* in_sizes, int n_in,
                              void* d_out, int out_size, void* d_ws, size_t ws_size,
                              hipStream_t stream) {
  (void)in_sizes; (void)n_in; (void)out_size; (void)ws_size;
  const float* gc   = (const float*)d_in[0];
  const int*   slen = (const int*)d_in[1];
  const float* W1 = (const float*)d_in[2];   const float* b1 = (const float*)d_in[3];
  const float* W2 = (const float*)d_in[4];   const float* b2 = (const float*)d_in[5];
  const float* W3 = (const float*)d_in[6];   const float* b3 = (const float*)d_in[7];
  const float* Wih1 = (const float*)d_in[8];  const float* bih1 = (const float*)d_in[9];
  const float* Whh1 = (const float*)d_in[10]; const float* bhh1 = (const float*)d_in[11];
  const float* Wih2 = (const float*)d_in[12]; const float* bih2 = (const float*)d_in[13];
  const float* Whh2 = (const float*)d_in[14]; const float* bhh2 = (const float*)d_in[15];
  const float* Wq1 = (const float*)d_in[16];  const float* bq1 = (const float*)d_in[17];
  const float* Wq2 = (const float*)d_in[18];  const float* bq2 = (const float*)d_in[19];
  const float* Wq3 = (const float*)d_in[20];  const float* bq3 = (const float*)d_in[21];
  const float* Wp1 = (const float*)d_in[22];  const float* bp1 = (const float*)d_in[23];
  const float* Wp2 = (const float*)d_in[24];  const float* bp2 = (const float*)d_in[25];
  const float* Wd1 = (const float*)d_in[26];  const float* bd1 = (const float*)d_in[27];
  const float* Wd2 = (const float*)d_in[28];  const float* bd2 = (const float*)d_in[29];
  const float* Wc1 = (const float*)d_in[30];  const float* bc1 = (const float*)d_in[31];
  const float* Wc2 = (const float*)d_in[32];  const float* bc2 = (const float*)d_in[33];

  // Output layout (flat, return order)
  float* out   = (float*)d_out;
  float* o_seq = out;                               // (B,32,128)
  float* o_q   = out + (size_t)BB * TT * HH;        // (B,1)
  float* o_p   = o_q + BB;                          // (B,8)
  float* o_d   = o_p + (size_t)BB * 8;              // (B,1)
  float* o_c   = o_d + BB;                          // (B,1)
  float* o_ctx = o_c + BB;                          // (B,64)

  // Workspace layout (~4.94 MB): bf16 x0 + bf16 weights + fused biases
  char* ws = (char*)d_ws;
  u16* x0 = (u16*)ws;                               // B*128 bf16
  u16* wb = (u16*)(ws + (size_t)BB * 128 * 2);
  u16* wW1   = wb;               // 256*128
  u16* wW2   = wW1 + 32768;      // 128*256
  u16* wW3   = wW2 + 32768;      // 64*128
  u16* wWih1 = wW3 + 8192;       // 512*128
  u16* wWhh1 = wWih1 + 65536;
  u16* wWih2 = wWhh1 + 65536;
  u16* wWhh2 = wWih2 + 65536;
  u16* wWq1  = wWhh2 + 65536;    // 128*192
  u16* wWq2  = wWq1 + 24576;     // 64*128
  float* bsum1 = (float*)(wWq2 + 8192);
  float* bsum2 = bsum1 + 512;

  // Weight prep (all weights ~1.3 MB -> L2 resident afterwards)
#define CVT(s, d, n) convert_bf16_kernel<<<((n) + 255) / 256, 256, 0, stream>>>((s), (d), (n))
  CVT(W1, wW1, 32768); CVT(W2, wW2, 32768); CVT(W3, wW3, 8192);
  CVT(Wih1, wWih1, 65536); CVT(Whh1, wWhh1, 65536);
  CVT(Wih2, wWih2, 65536); CVT(Whh2, wWhh2, 65536);
  CVT(Wq1, wWq1, 24576); CVT(Wq2, wWq2, 8192);
#undef CVT
  bias_sum_kernel<<<2, 256, 0, stream>>>(bih1, bhh1, bsum1, 512);
  bias_sum_kernel<<<2, 256, 0, stream>>>(bih2, bhh2, bsum2, 512);

  encoder_kernel<<<BB / 32, 64, 0, stream>>>(gc, wW1, b1, wW2, b2, wW3, b3,
                                             o_ctx, x0);
  heads_kernel<<<BB / 256, 256, 0, stream>>>(o_ctx, Wp1, bp1, Wp2, bp2,
                                             Wd1, bd1, Wd2, bd2,
                                             Wc1, bc1, Wc2, bc2,
                                             o_p, o_d, o_c);
  // LDS: x0(16K) + h1 dbl(32K) + h2 dbl(32K) = 80 KB (c-state in VGPRs)
  const size_t lsmem = (size_t)5 * 64 * 128 * 2;
  lstm_kernel<<<BB / 64, 512, lsmem, stream>>>(x0, wWih1, wWhh1, wWih2, wWhh2,
                                               bsum1, bsum2, slen, o_seq);
  quality_kernel<<<BB / 32, 64, 0, stream>>>(o_seq, o_ctx, wWq1, bq1, wWq2, bq2,
                                             Wq3, bq3, slen, o_q);
}
